// MSAModule_1898375545295
// MI455X (gfx1250) — compile-verified, matrix operand order checked by reading, not run on hardware
//
#include <hip/hip_runtime.h>
#include <cstdint>
#include <cstddef>

// =====================================================================
// MSAModule forward for gfx1250 (MI455X).
// All GEMM-like contractions run through one stride-generic batched
// WMMA kernel using v_wmma_f32_16x16x32_bf16 (f32 accum, bf16 operands
// staged in LDS, float4 vectorized staging on unit-stride operands).
// Everything else (LN / softmax / gating) is wave32 VALU code.
// z accumulates in-place in d_out.
//
// Input pointer order = setup_inputs() dict order; 'params' is a nested
// dict flattened in JAX pytree (sorted-key) order -- see enum below.
// Workspace requirement: ~575 MB of float scratch.
// =====================================================================

typedef long long i64;
typedef __attribute__((ext_vector_type(16))) __bf16 v16bf_t;
typedef __attribute__((ext_vector_type(8)))  float  v8f_t;

#define TB 128   // threads per GEMM block (4 waves)
#define BM 64
#define BN 64
#define BK 32
#define KP 40    // padded LDS K-stride (halves)

// ------------------------- input indices ------------------------------
enum {
  IN_Z = 0, IN_EMB, IN_MSA, IN_HASDEL, IN_DELVAL, IN_MSAMASK, IN_PADMASK,
  // params: layers[0] first (dict keys sorted), then msa_proj, s_proj
  P_MT_FC1 = 7, P_MT_FC2, P_MT_FC3, P_MT_LNB, P_MT_LNG,                 // msa_trans
  P_OPM_LNB, P_OPM_LNG, P_OPM_PA, P_OPM_PB, P_OPM_POB, P_OPM_POW,       // opm
  P_PWA_LNMB, P_PWA_LNMG, P_PWA_LNZB, P_PWA_LNZG,
  P_PWA_PG, P_PWA_PM, P_PWA_PO, P_PWA_PZ,                               // pwa
  P_TAE_BO, P_TAE_LNB, P_TAE_LNG, P_TAE_WB, P_TAE_WG, P_TAE_WK,
  P_TAE_WO, P_TAE_WQ, P_TAE_WV,                                         // tae
  P_TAS_BO, P_TAS_LNB, P_TAS_LNG, P_TAS_WB, P_TAS_WG, P_TAS_WK,
  P_TAS_WO, P_TAS_WQ, P_TAS_WV,                                         // tas
  P_TMI_GIN, P_TMI_GOUT, P_TMI_LNINB, P_TMI_LNING, P_TMI_LNOUTB,
  P_TMI_LNOUTG, P_TMI_PIN, P_TMI_POUT,                                  // tmi
  P_TMO_GIN, P_TMO_GOUT, P_TMO_LNINB, P_TMO_LNING, P_TMO_LNOUTB,
  P_TMO_LNOUTG, P_TMO_PIN, P_TMO_POUT,                                  // tmo
  P_ZT_FC1, P_ZT_FC2, P_ZT_FC3, P_ZT_LNB, P_ZT_LNG,                     // z_trans
  P_MSA_PROJ, P_S_PROJ
};

// ------------------------- GEMM kernel --------------------------------
// C[b2,b1] = alpha * A[b2,b1] @ B[b2,b1]  (+ bias[n]) (+ C if accumulate)
// Fully stride-generic: element (m,k) of A is A[m*sAm + k*sAk], etc.
struct GemmP {
  const float* A; const float* B; float* C; const float* bias;
  int M, N, K;
  i64 sAm, sAk, sBk, sBn, sCm, sCn;
  i64 bA1, bA2, bB1, bB2, bC1, bC2;
  int nb1;
  float alpha;
  int accumulate;
};

__global__ __launch_bounds__(TB) void k_gemm(GemmP p) {
  __shared__ __bf16 As[BM][KP];  // [m][k]
  __shared__ __bf16 Bs[BN][KP];  // [n][k]  (B tile stored transposed)

  const int bz = blockIdx.z;
  const int b1 = bz % p.nb1, b2 = bz / p.nb1;
  const float* A = p.A + (i64)b1 * p.bA1 + (i64)b2 * p.bA2;
  const float* B = p.B + (i64)b1 * p.bB1 + (i64)b2 * p.bB2;
  float*       C = p.C + (i64)b1 * p.bC1 + (i64)b2 * p.bC2;

  const int m0 = blockIdx.y * BM;
  const int n0 = blockIdx.x * BN;
  const int tid  = threadIdx.x;
  const int lane = tid & 31;
  const int wave = tid >> 5;
  const int wm = (wave >> 1) * 32;  // wave sub-tile origin
  const int wn = (wave & 1) * 32;

  // staging path selection (uniform across the block)
  const bool fastA  = (p.sAk == 1) && ((p.sAm & 3) == 0);
  const bool fastBk = (p.sBk == 1) && ((p.sBn & 3) == 0);
  const bool fastBn = (p.sBn == 1) && ((p.sBk & 3) == 0);

  v8f_t acc[2][2];
#pragma unroll
  for (int a = 0; a < 2; ++a)
#pragma unroll
    for (int b = 0; b < 2; ++b)
#pragma unroll
      for (int r = 0; r < 8; ++r) acc[a][b][r] = 0.f;

  const int half = lane >> 4;     // 0: lanes 0-15, 1: lanes 16-31
  const int l15  = lane & 15;
  const int klo  = half * 8;

  for (int k0 = 0; k0 < p.K; k0 += BK) {
    // ---- stage A tile (bf16, zero-padded at edges) ----
    if (fastA) {
      for (int t = tid; t < (BM * BK) / 4; t += TB) {
        int mm = t >> 3;
        int kk = (t & 7) * 4;
        int gm = m0 + mm, gk = k0 + kk;
        if (gm < p.M && gk + 3 < p.K) {
          const float4 f = *(const float4*)(A + (i64)gm * p.sAm + gk);
          As[mm][kk + 0] = (__bf16)f.x; As[mm][kk + 1] = (__bf16)f.y;
          As[mm][kk + 2] = (__bf16)f.z; As[mm][kk + 3] = (__bf16)f.w;
          // speculative prefetch of next K tile (silently dropped if OOB)
          __builtin_prefetch((const void*)(A + (i64)gm * p.sAm + gk + BK), 0, 0);
        } else {
#pragma unroll
          for (int u = 0; u < 4; ++u) {
            int gku = gk + u;
            int ok = (gm < p.M) & (gku < p.K);
            i64 idx = (i64)(ok ? gm : 0) * p.sAm + (ok ? gku : 0);
            float v = A[idx];
            As[mm][kk + u] = (__bf16)(ok ? v : 0.f);
          }
        }
      }
    } else {
      for (int t = tid; t < BM * BK; t += TB) {
        int mm = t >> 5, kk = t & 31;
        int gm = m0 + mm, gk = k0 + kk;
        int ok = (gm < p.M) & (gk < p.K);
        i64 idx = (i64)(ok ? gm : 0) * p.sAm + (i64)(ok ? gk : 0) * p.sAk;
        float v = A[idx];
        As[mm][kk] = (__bf16)(ok ? v : 0.f);
      }
    }

    // ---- stage B tile transposed: Bs[n][k] ----
    if (fastBk) {
      for (int t = tid; t < (BN * BK) / 4; t += TB) {
        int nn = t >> 3;
        int kk = (t & 7) * 4;
        int gn = n0 + nn, gk = k0 + kk;
        if (gn < p.N && gk + 3 < p.K) {
          const float4 f = *(const float4*)(B + (i64)gn * p.sBn + gk);
          Bs[nn][kk + 0] = (__bf16)f.x; Bs[nn][kk + 1] = (__bf16)f.y;
          Bs[nn][kk + 2] = (__bf16)f.z; Bs[nn][kk + 3] = (__bf16)f.w;
        } else {
#pragma unroll
          for (int u = 0; u < 4; ++u) {
            int gku = gk + u;
            int ok = (gn < p.N) & (gku < p.K);
            i64 idx = (i64)(ok ? gn : 0) * p.sBn + (ok ? gku : 0);
            float v = B[idx];
            Bs[nn][kk + u] = (__bf16)(ok ? v : 0.f);
          }
        }
      }
    } else if (fastBn) {
      for (int t = tid; t < (BN * BK) / 4; t += TB) {
        int kk = t >> 4;
        int nn = (t & 15) * 4;
        int gk = k0 + kk, gn = n0 + nn;
        if (gk < p.K && gn + 3 < p.N) {
          const float4 f = *(const float4*)(B + (i64)gk * p.sBk + gn);
          Bs[nn + 0][kk] = (__bf16)f.x; Bs[nn + 1][kk] = (__bf16)f.y;
          Bs[nn + 2][kk] = (__bf16)f.z; Bs[nn + 3][kk] = (__bf16)f.w;
        } else {
#pragma unroll
          for (int u = 0; u < 4; ++u) {
            int gnu = gn + u;
            int ok = (gnu < p.N) & (gk < p.K);
            i64 idx = (i64)(ok ? gk : 0) * p.sBk + (ok ? gnu : 0);
            float v = B[idx];
            Bs[nn + u][kk] = (__bf16)(ok ? v : 0.f);
          }
        }
      }
    } else {
      for (int t = tid; t < BN * BK; t += TB) {
        int nn = t >> 5, kk = t & 31;
        int gn = n0 + nn, gk = k0 + kk;
        int ok = (gn < p.N) & (gk < p.K);
        i64 idx = (i64)(ok ? gk : 0) * p.sBk + (i64)(ok ? gn : 0) * p.sBn;
        float v = B[idx];
        Bs[nn][kk] = (__bf16)(ok ? v : 0.f);
      }
    }
    __syncthreads();

    v16bf_t af[2], bfr[2];
#pragma unroll
    for (int mt = 0; mt < 2; ++mt) {
      int row = wm + mt * 16 + l15;
#pragma unroll
      for (int e = 0; e < 16; ++e) {
        int kk = klo + ((e < 8) ? e : (8 + e));   // K layout of 16-bit A frags
        af[mt][e] = As[row][kk];
      }
    }
#pragma unroll
    for (int nt = 0; nt < 2; ++nt) {
      int col = wn + nt * 16 + l15;
#pragma unroll
      for (int e = 0; e < 16; ++e) {
        int kk = klo + ((e < 8) ? e : (8 + e));
        bfr[nt][e] = Bs[col][kk];
      }
    }
#pragma unroll
    for (int mt = 0; mt < 2; ++mt)
#pragma unroll
      for (int nt = 0; nt < 2; ++nt)
        acc[mt][nt] = __builtin_amdgcn_wmma_f32_16x16x32_bf16(
            false, af[mt], false, bfr[nt], (short)0, acc[mt][nt], false, false);
    __syncthreads();
  }

  // epilogue: D layout -- lane(0-15): N=lane, M=r ; lane(16-31): N=lane-16, M=8+r
#pragma unroll
  for (int mt = 0; mt < 2; ++mt)
#pragma unroll
    for (int nt = 0; nt < 2; ++nt)
#pragma unroll
      for (int r = 0; r < 8; ++r) {
        int gm = m0 + wm + mt * 16 + half * 8 + r;
        int gn = n0 + wn + nt * 16 + l15;
        if (gm < p.M && gn < p.N) {
          i64 idx = (i64)gm * p.sCm + (i64)gn * p.sCn;
          float v = p.alpha * acc[mt][nt][r];
          if (p.bias) v += p.bias[gn];
          if (p.accumulate) v += C[idx];
          C[idx] = v;
        }
      }
}

// ------------------------- LayerNorm ----------------------------------
__global__ void k_layernorm(const float* __restrict__ x, float* __restrict__ y,
                            const float* __restrict__ g, const float* __restrict__ b,
                            int rows, int D) {
  int wid = blockIdx.x * (blockDim.x >> 5) + (threadIdx.x >> 5);
  if (wid >= rows) return;
  int lane = threadIdx.x & 31;
  const float* xr = x + (size_t)wid * D;
  float s = 0.f, s2 = 0.f;
  for (int c = lane; c < D; c += 32) { float v = xr[c]; s += v; s2 += v * v; }
  for (int o = 16; o; o >>= 1) { s += __shfl_xor(s, o, 32); s2 += __shfl_xor(s2, o, 32); }
  float mu = s / D;
  float var = s2 / D - mu * mu;
  float r = rsqrtf(var + 1e-5f);
  float* yr = y + (size_t)wid * D;
  for (int c = lane; c < D; c += 32) yr[c] = (xr[c] - mu) * r * g[c] + b[c];
}

// ------------------------- Softmax kernels ----------------------------
// PWA: bmat [N*N, 8] -> w [8, N, N], softmax over j with pair-mask bias
__global__ void k_pwa_softmax(const float* __restrict__ bmat, const float* __restrict__ maskp,
                              float* __restrict__ w, int H, int Nn) {
  int wid = blockIdx.x * (blockDim.x >> 5) + (threadIdx.x >> 5);
  if (wid >= H * Nn) return;
  int lane = threadIdx.x & 31;
  int h = wid / Nn, i = wid % Nn;
  const int cnt = Nn / 32;   // == 8
  float v[8];
  float mx = -1e30f;
  for (int t = 0; t < cnt; ++t) {
    int j = t * 32 + lane;
    float x = bmat[((size_t)i * Nn + j) * 8 + h] +
              (1.f - maskp[(size_t)i * Nn + j]) * -1e6f;
    v[t] = x; mx = fmaxf(mx, x);
  }
  for (int o = 16; o; o >>= 1) mx = fmaxf(mx, __shfl_xor(mx, o, 32));
  float s = 0.f;
  for (int t = 0; t < cnt; ++t) { v[t] = __expf(v[t] - mx); s += v[t]; }
  for (int o = 16; o; o >>= 1) s += __shfl_xor(s, o, 32);
  float inv = 1.f / s;
  for (int t = 0; t < cnt; ++t)
    w[(size_t)h * Nn * Nn + (size_t)i * Nn + t * 32 + lane] = v[t] * inv;
}

// tri-attention: in-place softmax over k on logits chunk [CI, H, N, N]
// with +trib[h,q,k] and +1e9*(mask[i,k]-1)
__global__ void k_tri_softmax(float* __restrict__ lg, const float* __restrict__ trib,
                              const float* __restrict__ maskp,
                              int i0, int CI, int H, int Nn) {
  int wid = blockIdx.x * (blockDim.x >> 5) + (threadIdx.x >> 5);
  if (wid >= CI * H * Nn) return;
  int lane = threadIdx.x & 31;
  int q = wid % Nn;
  int rest = wid / Nn;
  int h = rest % H, il = rest / H;
  int ig = i0 + il;
  float* row = lg + (((size_t)il * H + h) * Nn + q) * Nn;
  const int cnt = Nn / 32;
  float v[8];
  float mx = -1e30f;
  for (int t = 0; t < cnt; ++t) {
    int k = t * 32 + lane;
    float x = row[k] + trib[((size_t)q * Nn + k) * H + h] +
              1e9f * (maskp[(size_t)ig * Nn + k] - 1.f);
    v[t] = x; mx = fmaxf(mx, x);
  }
  for (int o = 16; o; o >>= 1) mx = fmaxf(mx, __shfl_xor(mx, o, 32));
  float s = 0.f;
  for (int t = 0; t < cnt; ++t) { v[t] = __expf(v[t] - mx); s += v[t]; }
  for (int o = 16; o; o >>= 1) s += __shfl_xor(s, o, 32);
  float inv = 1.f / s;
  for (int t = 0; t < cnt; ++t) row[t * 32 + lane] = v[t] * inv;
}

// ------------------------- elementwise kernels ------------------------
__device__ __forceinline__ float sigf(float x) { return 1.f / (1.f + __expf(-x)); }
#define GS_LOOP(n) for (size_t i = (size_t)blockIdx.x * blockDim.x + threadIdx.x; \
                        i < (n); i += (size_t)gridDim.x * blockDim.x)

__global__ void k_sigmoid_mul(float* out, const float* g, const float* x, size_t n) {
  GS_LOOP(n) out[i] = sigf(g[i]) * x[i];
}
__global__ void k_silu_mul(float* out, const float* a, const float* b, size_t n) {
  GS_LOOP(n) { float av = a[i]; out[i] = av * sigf(av) * b[i]; }
}
__global__ void k_gate_add(float* z, const float* a, const float* g, size_t n) {
  GS_LOOP(n) z[i] += a[i] * sigf(g[i]);
}
__global__ void k_gate_pair(float* out, const float* p, const float* g,
                            const float* maskr, size_t n, int C) {
  GS_LOOP(n) out[i] = p[i] * sigf(g[i]) * maskr[i / C];
}
__global__ void k_mask_rows(float* x, const float* maskr, size_t n, int C) {
  GS_LOOP(n) x[i] *= maskr[i / C];
}
__global__ void k_add_rowbrd(float* x, const float* e, size_t n, int C, int Nt) {
  GS_LOOP(n) { size_t r = i / C; int c = (int)(i % C); int nn = (int)(r % Nt);
               x[i] += e[(size_t)nn * C + c]; }
}
__global__ void k_build_mcat(float* out, const float* msa, const float* hd,
                             const float* dv, size_t rows) {
  size_t n = rows * 35;
  GS_LOOP(n) {
    size_t r = i / 35; int t = (int)(i % 35);
    out[i] = (t < 33) ? msa[r * 33 + t] : (t == 33 ? hd[r] : dv[r]);
  }
}
__global__ void k_token_mask(float* maskp, const float* pad, int Nt) {
  size_t n = (size_t)Nt * Nt;
  GS_LOOP(n) maskp[i] = pad[i / Nt] * pad[i % Nt];
}
__global__ void k_opm_fin(float* z, const float* u, const float* pob,
                          const float* num, size_t n) {
  GS_LOOP(n) z[i] += (u[i] + pob[i % 128]) / fmaxf(num[i / 128], 1.f);
}
__global__ void k_opm_transpose_a(float* a2, const float* a, int Sm, int Nt) {
  size_t n = (size_t)Sm * Nt * 32;
  GS_LOOP(n) {
    int c = (int)(i % 32); size_t r = i / 32;
    int ii = (int)(r % Nt); int s = (int)(r / Nt);
    a2[((size_t)ii * 32 + c) * Sm + s] = a[i];
  }
}
// oc [(il*32+c), (j*32+d)] -> pc [(il*N+j), c*32+d]   (CI = 32)
__global__ void k_opm_permute(float* pc, const float* oc, int Nt) {
  size_t n = (size_t)32 * Nt * 1024;
  GS_LOOP(n) {
    int cd = (int)(i % 1024); size_t row = i / 1024;
    int j = (int)(row % Nt); int il = (int)(row / Nt);
    int c = cd >> 5, d = cd & 31;
    pc[i] = oc[((size_t)il * 32 + c) * ((size_t)Nt * 32) + (size_t)j * 32 + d];
  }
}
__global__ void k_transpose_z(float* zt, const float* z, int Nt) {
  size_t n = (size_t)Nt * Nt * 128;
  GS_LOOP(n) {
    int c = (int)(i % 128); size_t r = i / 128;
    int j = (int)(r % Nt); int ii = (int)(r / Nt);
    zt[((size_t)j * Nt + ii) * 128 + c] = z[i];
  }
}
__global__ void k_add_transposed(float* z, const float* u, int Nt) {
  size_t n = (size_t)Nt * Nt * 128;
  GS_LOOP(n) {
    int c = (int)(i % 128); size_t r = i / 128;
    int j = (int)(r % Nt); int ii = (int)(r / Nt);
    z[i] += u[((size_t)j * Nt + ii) * 128 + c];
  }
}

// ------------------------- host helpers -------------------------------
static inline unsigned gsz(size_t n) {
  size_t g = (n + 255) / 256;
  return (unsigned)(g > 1048576 ? 1048576 : g);
}

static void gemm(hipStream_t st, const float* A, const float* B, float* C,
                 int M, int N, int K,
                 i64 sAm, i64 sAk, i64 sBk, i64 sBn, i64 sCm, i64 sCn,
                 float alpha, int accum, const float* bias,
                 int nb1, int nb2,
                 i64 bA1, i64 bA2, i64 bB1, i64 bB2, i64 bC1, i64 bC2) {
  GemmP p{A, B, C, bias, M, N, K, sAm, sAk, sBk, sBn, sCm, sCn,
          bA1, bA2, bB1, bB2, bC1, bC2, nb1, alpha, accum};
  dim3 grid((N + BN - 1) / BN, (M + BM - 1) / BM, (unsigned)(nb1 * nb2));
  k_gemm<<<grid, dim3(TB), 0, st>>>(p);
}
static void gemm_rm(hipStream_t st, const float* A, const float* B, float* C,
                    int M, int N, int K, float alpha = 1.f, int accum = 0,
                    const float* bias = nullptr) {
  gemm(st, A, B, C, M, N, K, K, 1, N, 1, N, 1, alpha, accum, bias,
       1, 1, 0, 0, 0, 0, 0, 0);
}
static void layernorm(hipStream_t st, const float* x, float* y,
                      const float* g, const float* b, int rows, int D) {
  k_layernorm<<<(rows + 7) / 8, 256, 0, st>>>(x, y, g, b, rows, D);
}

// pr = {g_in, g_out, ln_in_b, ln_in_g, ln_out_b, ln_out_g, p_in, p_out}
static void tri_mul_run(hipStream_t st, float* zio, const float* const* pr,
                        const float* maskp, float* zn,
                        float* buf1, float* buf2, float* buf3, int outgoing) {
  const int Nt = 256, NN = Nt * Nt;
  layernorm(st, zio, zn, pr[3], pr[2], NN, 128);
  gemm_rm(st, zn, pr[6], buf1, NN, 256, 128);  // p_in
  gemm_rm(st, zn, pr[0], buf2, NN, 256, 128);  // g_in
  k_gate_pair<<<gsz((size_t)NN * 256), 256, 0, st>>>(buf1, buf1, buf2, maskp,
                                                     (size_t)NN * 256, 256);
  const float* Aa = buf1;         // channels 0..127
  const float* Bb = buf1 + 128;   // channels 128..255
  if (outgoing) {
    // t[i,j,d] = sum_k a[i,k,d]*b[j,k,d] ; batch over d
    gemm(st, Aa, Bb, buf2, Nt, Nt, Nt,
         (i64)Nt * 256, 256, 256, (i64)Nt * 256, (i64)Nt * 128, 128,
         1.f, 0, nullptr, 128, 1, 1, 0, 1, 0, 1, 0);
  } else {
    // t[i,j,d] = sum_k a[k,i,d]*b[k,j,d]
    gemm(st, Aa, Bb, buf2, Nt, Nt, Nt,
         256, (i64)Nt * 256, (i64)Nt * 256, 256, (i64)Nt * 128, 128,
         1.f, 0, nullptr, 128, 1, 1, 0, 1, 0, 1, 0);
  }
  layernorm(st, buf2, buf3, pr[5], pr[4], NN, 128);
  gemm_rm(st, buf3, pr[7], buf1, NN, 128, 128);  // p_out -> tp
  gemm_rm(st, zn, pr[1], buf2, NN, 128, 128);    // g_out
  k_gate_add<<<gsz((size_t)NN * 128), 256, 0, st>>>(zio, buf1, buf2,
                                                    (size_t)NN * 128);
}

// pr = {bo, ln_b, ln_g, wb, wg, wk, wo, wq, wv}
static void tri_att_run(hipStream_t st, const float* zsrc, const float* const* pr,
                        const float* maskp, float* cout, int accum,
                        float* zn, float* buf1, float* buf2, float* buf3,
                        float* trib) {
  const int Nt = 256, NN = Nt * Nt, H = 4, CI = 32;
  layernorm(st, zsrc, zn, pr[2], pr[1], NN, 128);
  float* qb = buf1;
  float* kb = buf1 + (size_t)NN * 128;
  float* vb = buf1 + (size_t)2 * NN * 128;
  float* gb = buf1 + (size_t)3 * NN * 128;
  gemm_rm(st, zn, pr[7], qb, NN, 128, 128);   // wq
  gemm_rm(st, zn, pr[5], kb, NN, 128, 128);   // wk
  gemm_rm(st, zn, pr[8], vb, NN, 128, 128);   // wv
  gemm_rm(st, zn, pr[4], gb, NN, 128, 128);   // wg
  gemm_rm(st, zn, pr[3], trib, NN, H, 128);   // wb
  const float scale = 0.17677669529663687f;   // 1/sqrt(32)
  for (int i0 = 0; i0 < Nt; i0 += CI) {
    // logits[il,h,q,k] = scale * q[i,q,h,:]·k[i,k,h,:]
    gemm(st, qb + (size_t)i0 * Nt * 128, kb + (size_t)i0 * Nt * 128, buf3,
         Nt, Nt, 32,
         128, 1, 1, 128, Nt, 1,
         scale, 0, nullptr,
         H, CI,
         32, (i64)Nt * 128, 32, (i64)Nt * 128,
         (i64)Nt * Nt, (i64)H * Nt * Nt);
    int rows = CI * H * Nt;
    k_tri_softmax<<<(rows + 7) / 8, 256, 0, st>>>(buf3, trib, maskp, i0, CI, H, Nt);
    // o[i,q,h,d] = sum_k a[il,h,q,k] * v[i,k,h,d]
    gemm(st, buf3, vb + (size_t)i0 * Nt * 128, buf2 + (size_t)i0 * Nt * 128,
         Nt, 32, Nt,
         Nt, 1, 128, 1, 128, 1,
         1.f, 0, nullptr,
         H, CI,
         (i64)Nt * Nt, (i64)H * Nt * Nt, 32, (i64)Nt * 128, 32, (i64)Nt * 128);
  }
  k_sigmoid_mul<<<gsz((size_t)NN * 128), 256, 0, st>>>(buf2, gb, buf2,
                                                       (size_t)NN * 128);
  gemm_rm(st, buf2, pr[6], cout, NN, 128, 128, 1.f, accum, pr[0]);  // wo + bo
}

// ------------------------- top level ----------------------------------
extern "C" void kernel_launch(void* const* d_in, const int* in_sizes, int n_in,
                              void* d_out, int out_size, void* d_ws, size_t ws_size,
                              hipStream_t stream) {
  (void)in_sizes; (void)n_in; (void)out_size; (void)ws_size;
  const int Nt = 256, Sm = 512;
  const int SN = Sm * Nt, NN = Nt * Nt;
  auto F = [&](int i) { return (const float*)d_in[i]; };
  float* ws = (float*)d_ws;
  size_t off = 0;
  auto alloc = [&](size_t n) { float* p = ws + off; off += n; return p; };
  float* m     = alloc((size_t)SN * 64);
  float* mn    = alloc((size_t)SN * 64);
  float* e64   = alloc((size_t)Nt * 64);
  float* buf1  = alloc((size_t)SN * 256);
  float* buf2  = alloc((size_t)SN * 256);
  float* buf3  = alloc((size_t)SN * 256);
  float* zn    = alloc((size_t)NN * 128);
  float* zupd  = alloc((size_t)NN * 128);
  float* ztb   = alloc((size_t)NN * 128);
  float* bmat  = alloc((size_t)NN * 8);
  float* wbuf  = alloc((size_t)8 * NN);
  float* trib  = alloc((size_t)NN * 4);
  float* numb  = alloc((size_t)NN);
  float* maskp = alloc((size_t)NN);
  float* z = (float*)d_out;

  // z accumulates in d_out
  hipMemcpyAsync(z, d_in[IN_Z], (size_t)NN * 128 * sizeof(float),
                 hipMemcpyDeviceToDevice, stream);
  k_token_mask<<<gsz((size_t)NN), 256, 0, stream>>>(maskp, F(IN_PADMASK), Nt);

  // ---- build m: [msa | has_del | del_val] @ msa_proj + emb @ s_proj ----
  k_build_mcat<<<gsz((size_t)SN * 35), 256, 0, stream>>>(buf1, F(IN_MSA),
                                                         F(IN_HASDEL), F(IN_DELVAL),
                                                         (size_t)SN);
  gemm_rm(stream, buf1, F(P_MSA_PROJ), m, SN, 64, 35);
  gemm_rm(stream, F(IN_EMB), F(P_S_PROJ), e64, Nt, 64, 384);
  k_add_rowbrd<<<gsz((size_t)SN * 64), 256, 0, stream>>>(m, e64, (size_t)SN * 64,
                                                         64, Nt);

  // ---- Pair-Weighted Averaging:  m += pwa(m, z) ----
  layernorm(stream, m, mn, F(P_PWA_LNMG), F(P_PWA_LNMB), SN, 64);
  layernorm(stream, z, zn, F(P_PWA_LNZG), F(P_PWA_LNZB), NN, 128);
  gemm_rm(stream, mn, F(P_PWA_PM), buf1, SN, 256, 64);   // v
  gemm_rm(stream, mn, F(P_PWA_PG), buf2, SN, 256, 64);   // gate pre-act
  gemm_rm(stream, zn, F(P_PWA_PZ), bmat, NN, 8, 128);    // pair bias
  k_pwa_softmax<<<(8 * Nt + 7) / 8, 256, 0, stream>>>(bmat, maskp, wbuf, 8, Nt);
  // o[s,i,h,d] = sum_j w[h,i,j] v[s,j,h,d]  (batch over h,s)
  gemm(stream, wbuf, buf1, buf3, Nt, 32, Nt,
       Nt, 1, 256, 1, 256, 1, 1.f, 0, nullptr,
       8, Sm, (i64)NN, 0, 32, (i64)Nt * 256, 32, (i64)Nt * 256);
  k_sigmoid_mul<<<gsz((size_t)SN * 256), 256, 0, stream>>>(buf1, buf2, buf3,
                                                           (size_t)SN * 256);
  gemm_rm(stream, buf1, F(P_PWA_PO), m, SN, 64, 256, 1.f, 1, nullptr);

  // ---- MSA transition:  m += trans(m) ----
  layernorm(stream, m, mn, F(P_MT_LNG), F(P_MT_LNB), SN, 64);
  gemm_rm(stream, mn, F(P_MT_FC1), buf1, SN, 256, 64);
  gemm_rm(stream, mn, F(P_MT_FC2), buf2, SN, 256, 64);
  k_silu_mul<<<gsz((size_t)SN * 256), 256, 0, stream>>>(buf1, buf1, buf2,
                                                        (size_t)SN * 256);
  gemm_rm(stream, buf1, F(P_MT_FC3), m, SN, 64, 256, 1.f, 1, nullptr);

  // ---- Outer Product Mean:  z += opm(m) ----
  layernorm(stream, m, mn, F(P_OPM_LNG), F(P_OPM_LNB), SN, 64);
  float* aop = buf1;
  float* bop = buf1 + (size_t)SN * 32;
  float* a2  = buf1 + (size_t)SN * 64;  // a transposed: [(i,c), s]
  gemm_rm(stream, mn, F(P_OPM_PA), aop, SN, 32, 64);
  gemm_rm(stream, mn, F(P_OPM_PB), bop, SN, 32, 64);
  k_mask_rows<<<gsz((size_t)SN * 32), 256, 0, stream>>>(aop, F(IN_MSAMASK),
                                                        (size_t)SN * 32, 32);
  k_mask_rows<<<gsz((size_t)SN * 32), 256, 0, stream>>>(bop, F(IN_MSAMASK),
                                                        (size_t)SN * 32, 32);
  k_opm_transpose_a<<<gsz((size_t)SN * 32), 256, 0, stream>>>(a2, aop, Sm, Nt);
  float* oc = buf2;  // outer chunk [(32*32), (N*32)]
  float* pc = buf3;  // permuted     [(32*N), (32*32)]
  for (int i0 = 0; i0 < Nt; i0 += 32) {
    gemm(stream, a2 + (size_t)i0 * 32 * Sm, bop, oc, 1024, Nt * 32, Sm,
         Sm, 1, (i64)Nt * 32, 1, (i64)Nt * 32, 1,
         1.f, 0, nullptr, 1, 1, 0, 0, 0, 0, 0, 0);
    k_opm_permute<<<gsz((size_t)32 * Nt * 1024), 256, 0, stream>>>(pc, oc, Nt);
    gemm_rm(stream, pc, F(P_OPM_POW), zupd + (size_t)i0 * Nt * 128,
            32 * Nt, 128, 1024);
  }
  // num[i,j] = sum_s mask[s,i]*mask[s,j]   (mask^T @ mask)
  gemm(stream, F(IN_MSAMASK), F(IN_MSAMASK), numb, Nt, Nt, Sm,
       1, Nt, Nt, 1, Nt, 1, 1.f, 0, nullptr, 1, 1, 0, 0, 0, 0, 0, 0);
  k_opm_fin<<<gsz((size_t)NN * 128), 256, 0, stream>>>(z, zupd, F(P_OPM_POB),
                                                       numb, (size_t)NN * 128);

  // ---- Triangle multiplicative updates ----
  {
    const float* pr[8] = {F(P_TMO_GIN), F(P_TMO_GOUT), F(P_TMO_LNINB),
                          F(P_TMO_LNING), F(P_TMO_LNOUTB), F(P_TMO_LNOUTG),
                          F(P_TMO_PIN), F(P_TMO_POUT)};
    tri_mul_run(stream, z, pr, maskp, zn, buf1, buf2, buf3, 1);
  }
  {
    const float* pr[8] = {F(P_TMI_GIN), F(P_TMI_GOUT), F(P_TMI_LNINB),
                          F(P_TMI_LNING), F(P_TMI_LNOUTB), F(P_TMI_LNOUTG),
                          F(P_TMI_PIN), F(P_TMI_POUT)};
    tri_mul_run(stream, z, pr, maskp, zn, buf1, buf2, buf3, 0);
  }

  // ---- Triangle attention (starting node) ----
  {
    const float* pr[9] = {F(P_TAS_BO), F(P_TAS_LNB), F(P_TAS_LNG), F(P_TAS_WB),
                          F(P_TAS_WG), F(P_TAS_WK), F(P_TAS_WO), F(P_TAS_WQ),
                          F(P_TAS_WV)};
    tri_att_run(stream, z, pr, maskp, z, 1, zn, buf1, buf2, buf3, trib);
  }
  // ---- Triangle attention (ending node): run on z^T, add back transposed ----
  {
    k_transpose_z<<<gsz((size_t)NN * 128), 256, 0, stream>>>(ztb, z, Nt);
    const float* pr[9] = {F(P_TAE_BO), F(P_TAE_LNB), F(P_TAE_LNG), F(P_TAE_WB),
                          F(P_TAE_WG), F(P_TAE_WK), F(P_TAE_WO), F(P_TAE_WQ),
                          F(P_TAE_WV)};
    tri_att_run(stream, ztb, pr, maskp, zupd, 0, zn, buf1, buf2, buf3, trib);
    k_add_transposed<<<gsz((size_t)NN * 128), 256, 0, stream>>>(z, zupd, Nt);
  }

  // ---- Pair transition:  z += trans(z) ----
  layernorm(stream, z, zn, F(P_ZT_LNG), F(P_ZT_LNB), NN, 128);
  gemm_rm(stream, zn, F(P_ZT_FC1), buf1, NN, 512, 128);
  gemm_rm(stream, zn, F(P_ZT_FC2), buf2, NN, 512, 128);
  k_silu_mul<<<gsz((size_t)NN * 512), 256, 0, stream>>>(buf1, buf1, buf2,
                                                        (size_t)NN * 512);
  gemm_rm(stream, buf1, F(P_ZT_FC3), z, NN, 128, 512, 1.f, 1, nullptr);
}
